// LPDNetOrign_80358838108662
// MI455X (gfx1250) — compile-verified, matrix operand order checked by reading, not run on hardware
//
#include <hip/hip_runtime.h>
#include <hip/hip_bf16.h>
#include <cstdint>
#include <cstddef>

// ---------------------------------------------------------------------------
// LPDNet (DGCNN-style) forward for MI455X / gfx1250, wave32 + bf16 WMMA.
// B=8, N=4096, K=20.  All channel-mixing on v_wmma_f32_16x16x32_bf16.
// All intermediate feature tensors are stored bf16 so WMMA fragments load as
// two global_load_b128 per operand (CDNA5 16-bit fragment = two contiguous
// 8-element K-chunks per lane). Accumulation, BN+LeakyReLU epilogues, top-k
// selection and the final output stay in f32.
// ---------------------------------------------------------------------------

typedef __attribute__((ext_vector_type(16))) __bf16 v16bf;
typedef __attribute__((ext_vector_type(8)))  __bf16 v8bf;
typedef __attribute__((ext_vector_type(8)))  float  v8f;

#define NPTS   4096
#define BATCH  8
#define KNN_K  20
#define NSLOPE 0.01f

// ---- WMMA fragment helpers -------------------------------------------------
// 16-bit A/B fragment layout (CDNA5 ISA 7.12.2): lane L holds row/col (L&15),
// K elements e=0..15 map to K = (e<8 ? e : e+8) + 8*(L>>4).  Pointers passed
// here must already include  base + 8*(lane>>4).

// bf16 source: two 16-byte vector loads (K-chunks at +0 and +16 elements)
__device__ __forceinline__ v16bf ld_frag_bf(const __bf16* __restrict__ p) {
  const v8bf c0 = *(const v8bf*)(p);
  const v8bf c1 = *(const v8bf*)(p + 16);
  v16bf a;
#pragma unroll
  for (int e = 0; e < 8; ++e) { a[e] = c0[e]; a[e + 8] = c1[e]; }
  return a;
}

// f32 source (LDS staging tile): scalar loads + convert
__device__ __forceinline__ v16bf ld_frag_f32(const float* __restrict__ p) {
  v16bf a;
#pragma unroll
  for (int e = 0; e < 16; ++e) {
    const int k = (e < 8) ? e : (e + 8);
    a[e] = (__bf16)p[k];
  }
  return a;
}

__device__ __forceinline__ v16bf zero_frag() {
  v16bf a;
#pragma unroll
  for (int e = 0; e < 16; ++e) a[e] = (__bf16)0.0f;
  return a;
}

__device__ __forceinline__ float leaky(float v) {
  return (v > 0.0f) ? v : NSLOPE * v;
}

// ---- conv1: (B,N,3) f32 -> (B,N,64) bf16 (K=3 too small for WMMA) ----------
__global__ void k_conv1(const float* __restrict__ x, const float* __restrict__ w,
                        const float* __restrict__ g, const float* __restrict__ b,
                        __bf16* __restrict__ h1) {
  const int t = blockIdx.x * blockDim.x + threadIdx.x;
  if (t >= BATCH * NPTS * 64) return;
  const int p = t >> 6, o = t & 63;
  const float* xp = x + (size_t)p * 3;
  const float* wo = w + (size_t)o * 3;
  float v = xp[0] * wo[0] + xp[1] * wo[1] + xp[2] * wo[2];
  h1[t] = (__bf16)leaky(v * g[o] + b[o]);
}

// ---- per-point squared norm from bf16 features -----------------------------
__global__ void k_xx(const __bf16* __restrict__ h, float* __restrict__ xx) {
  const int p = blockIdx.x * blockDim.x + threadIdx.x;
  if (p >= BATCH * NPTS) return;
  const __bf16* hp = h + (size_t)p * 64;
  float s = 0.f;
#pragma unroll
  for (int c = 0; c < 64; ++c) { const float v = (float)hp[c]; s += v * v; }
  xx[p] = s;
}

// ---- weight preparation: f32 -> compact bf16 -------------------------------
__global__ void k_cvt(const float* __restrict__ src, __bf16* __restrict__ dst,
                      const int n) {
  const int i = blockIdx.x * blockDim.x + threadIdx.x;
  if (i < n) dst[i] = (__bf16)src[i];
}

// split Wdg1 (64x128, edge = [ctr, nbr-ctr]) into Wdiff = Wa - Wb and Wb
__global__ void k_wprep(const float* __restrict__ w, __bf16* __restrict__ wdiff,
                        __bf16* __restrict__ wb) {
  const int t = blockIdx.x * blockDim.x + threadIdx.x;
  if (t >= 64 * 64) return;
  const int o = t >> 6, c = t & 63;
  const float a = w[o * 128 + c], bb = w[o * 128 + 64 + c];
  wdiff[t] = (__bf16)(a - bb);
  wb[t]    = (__bf16)bb;
}

// ---- generic WMMA GEMM: Y[P,CO] = act(g,b)( X[P,CI] @ W[CO,CI]^T ) ---------
// 128 threads = 4 waves; each wave owns one 16-row tile, loops over col tiles.
// OUTF32 selects f32 output (base vector / final), else bf16.
// TRANS stores conv5 output in (B, CO, N) layout.
template <int CI, int CO, bool ACT, bool TRANS, bool OUTF32>
__global__ __launch_bounds__(128) void k_gemm(const __bf16* __restrict__ X,
                                              const __bf16* __restrict__ W,
                                              const float* __restrict__ g,
                                              const float* __restrict__ bi,
                                              void* __restrict__ Yv) {
  const int t = threadIdx.x;
  const int lane = t & 31, wid = t >> 5;
  const int l15 = lane & 15, half = lane >> 4;
  const int rt = blockIdx.x * 4 + wid;     // 16-row tile index
  const int row = rt * 16 + l15;

  constexpr int KS = CI / 32;
  v16bf af[KS];
#pragma unroll
  for (int ks = 0; ks < KS; ++ks)
    af[ks] = ld_frag_bf(X + (size_t)row * CI + ks * 32 + 8 * half);

#pragma unroll 1
  for (int ct = 0; ct < CO / 16; ++ct) {
    const int col = ct * 16 + l15;
    v8f acc = {};
#pragma unroll
    for (int ks = 0; ks < KS; ++ks) {
      v16bf bf = ld_frag_bf(W + (size_t)col * CI + ks * 32 + 8 * half);
      acc = __builtin_amdgcn_wmma_f32_16x16x32_bf16(false, af[ks], false, bf,
                                                    (short)0, acc, false, false);
    }
    float gg = 1.f, bb = 0.f;
    if constexpr (ACT) { gg = g[col]; bb = bi[col]; }
#pragma unroll
    for (int r = 0; r < 8; ++r) {
      float v = acc[r];
      if constexpr (ACT) v = leaky(v * gg + bb);
      const int m = rt * 16 + r + 8 * half;
      if constexpr (TRANS) {
        const int bb2 = m >> 12, n = m & (NPTS - 1);
        ((float*)Yv)[((size_t)bb2 * CO + col) * NPTS + n] = v;
      } else if constexpr (OUTF32) {
        ((float*)Yv)[(size_t)m * CO + col] = v;
      } else {
        ((__bf16*)Yv)[(size_t)m * CO + col] = (__bf16)v;
      }
    }
  }
}

// ---- fused Gram + top-K (descending, index tie-break = jax.lax.top_k) ------
// Block: 128 threads (4 waves) owns 16 query rows of one batch.
// CDIM==64: waves compute 16x16 Gram tiles with WMMA (bf16 feats), dist->LDS.
// CDIM==3 : distances computed scalar from f32 xyz.
template <int CDIM>
__global__ __launch_bounds__(128) void k_knn(const void* __restrict__ Fv,
                                             const float* __restrict__ colsq,
                                             int* __restrict__ idxout) {
  __shared__ float sdist[16][65];
  __shared__ float slv[128 * KNN_K];
  __shared__ int   sli[128 * KNN_K];

  const int t = threadIdx.x;
  const int lane = t & 31, wid = t >> 5;
  const int l15 = lane & 15, half = lane >> 4;
  const int b = blockIdx.x >> 8;            // 256 row-tiles per batch
  const int rt = blockIdx.x & 255;
  const int bbase = b << 12;

  float lv[KNN_K];
  int   li[KNN_K];
#pragma unroll
  for (int j = 0; j < KNN_K; ++j) { lv[j] = -INFINITY; li[j] = 0x7fffffff; }

  const int irow = t >> 3;                  // insertion row 0..15
  const int slot = t & 7;

  v16bf af[2];
  float px = 0.f, py = 0.f, pz = 0.f;
  if constexpr (CDIM == 64) {
    const __bf16* F = (const __bf16*)Fv;
    const int arow = bbase + rt * 16 + l15;
#pragma unroll
    for (int ks = 0; ks < 2; ++ks)
      af[ks] = ld_frag_bf(F + (size_t)arow * 64 + ks * 32 + 8 * half);
  } else {
    const float* F = (const float*)Fv;
    const float* pr = F + (size_t)(bbase + rt * 16 + irow) * 3;
    px = pr[0]; py = pr[1]; pz = pr[2];
  }

  for (int cb = 0; cb < NPTS; cb += 64) {
    if constexpr (CDIM == 64) {
      const __bf16* F = (const __bf16*)Fv;
      const int cl = wid * 16 + l15;
      const int cp = bbase + cb + cl;
      v8f acc = {};
#pragma unroll
      for (int ks = 0; ks < 2; ++ks) {
        v16bf bf = ld_frag_bf(F + (size_t)cp * 64 + ks * 32 + 8 * half);
        acc = __builtin_amdgcn_wmma_f32_16x16x32_bf16(false, af[ks], false, bf,
                                                      (short)0, acc, false, false);
      }
      const float cq = colsq[cp];
#pragma unroll
      for (int r = 0; r < 8; ++r)
        sdist[r + 8 * half][cl] = 2.0f * acc[r] - cq;   // row term is constant
      __syncthreads();
    }
#pragma unroll
    for (int s = 0; s < 8; ++s) {
      const int cl = slot + 8 * s;
      const int gcol = cb + cl;
      float val;
      if constexpr (CDIM == 64) {
        val = sdist[irow][cl];
      } else {
        const float* F = (const float*)Fv;
        const float* q = F + (size_t)(bbase + gcol) * 3;
        const float q0 = q[0], q1 = q[1], q2 = q[2];
        val = 2.0f * (px * q0 + py * q1 + pz * q2) - (q0 * q0 + q1 * q1 + q2 * q2);
      }
      if (val > lv[KNN_K - 1]) {            // strict: keeps earlier index on ties
        float cv = val; int ci = gcol;
#pragma unroll
        for (int j = 0; j < KNN_K; ++j) {
          const bool gt = cv > lv[j];
          const float tv = lv[j]; const int ti = li[j];
          lv[j] = gt ? cv : tv; li[j] = gt ? ci : ti;
          cv = gt ? tv : cv;    ci = gt ? ti : ci;
        }
      }
    }
    if constexpr (CDIM == 64) __syncthreads();
  }

#pragma unroll
  for (int j = 0; j < KNN_K; ++j) { slv[t * KNN_K + j] = lv[j]; sli[t * KNN_K + j] = li[j]; }
  __syncthreads();

  if (t < 16) {                             // stable 8-way merge of sorted lists
    int hd[8];
#pragma unroll
    for (int s = 0; s < 8; ++s) hd[s] = 0;
    const int outbase = (bbase + rt * 16 + t) * KNN_K;
    for (int j = 0; j < KNN_K; ++j) {
      float bv = -INFINITY; int bi2 = 0x7fffffff; int bs = -1;
#pragma unroll
      for (int s = 0; s < 8; ++s) {
        if (hd[s] < KNN_K) {
          const int off = (t * 8 + s) * KNN_K + hd[s];
          const float v = slv[off]; const int id = sli[off];
          if (v > bv || (v == bv && id < bi2)) { bv = v; bi2 = id; bs = s; }
        }
      }
#pragma unroll
      for (int s = 0; s < 8; ++s)
        if (s == bs) hd[s]++;
      idxout[outbase + j] = bi2;
    }
  }
}

// ---- fused edge-conv block: gather K nbrs -> GEMM1 -> GEMM2 -> max over k --
// 4 waves per block, one point per wave.
// y1[j] = act1(g1*((base?) + W1 . F[idx[j]]) + b1);
// out = max_j act2(g2*(W2 . y1[j]) + b2).  Neighbor rows j>=20 are padding.
template <bool HASBASE>
__global__ __launch_bounds__(128) void k_edge(const __bf16* __restrict__ F,
                                              const int* __restrict__ idx,
                                              const float* __restrict__ base,
                                              const __bf16* __restrict__ W1,
                                              const float* __restrict__ g1,
                                              const float* __restrict__ b1,
                                              const __bf16* __restrict__ W2,
                                              const float* __restrict__ g2,
                                              const float* __restrict__ b2,
                                              __bf16* __restrict__ out) {
  __shared__ float y1[4][32][65];
  const int lane = threadIdx.x & 31, wid = threadIdx.x >> 5;
  const int p = blockIdx.x * 4 + wid;
  const int bbase = (p >> 12) << 12;
  const int l15 = lane & 15, half = lane >> 4;
  const int* ip = idx + (size_t)p * KNN_K;

  // gather A fragments (2 row tiles x 2 k-steps); rows >= K are zero
  v16bf a1[2][2];
#pragma unroll
  for (int mt = 0; mt < 2; ++mt) {
    const int j = mt * 16 + l15;
    const bool valid = (j < KNN_K);
    const int nb = valid ? ip[j] : 0;
    const __bf16* src = F + (size_t)(bbase + nb) * 64;
#pragma unroll
    for (int ks = 0; ks < 2; ++ks)
      a1[mt][ks] = valid ? ld_frag_bf(src + ks * 32 + 8 * half) : zero_frag();
  }

#pragma unroll 1
  for (int ct = 0; ct < 4; ++ct) {
    const int col = ct * 16 + l15;
    v16bf bw[2];
#pragma unroll
    for (int ks = 0; ks < 2; ++ks)
      bw[ks] = ld_frag_bf(W1 + (size_t)col * 64 + ks * 32 + 8 * half);
    float bs = 0.f;
    if constexpr (HASBASE) bs = base[(size_t)p * 64 + col];
    const float gg = g1[col], bb = b1[col];
#pragma unroll
    for (int mt = 0; mt < 2; ++mt) {
      v8f acc = {};
#pragma unroll
      for (int ks = 0; ks < 2; ++ks)
        acc = __builtin_amdgcn_wmma_f32_16x16x32_bf16(false, a1[mt][ks], false, bw[ks],
                                                      (short)0, acc, false, false);
#pragma unroll
      for (int r = 0; r < 8; ++r)
        y1[wid][mt * 16 + r + 8 * half][col] = leaky((acc[r] + bs) * gg + bb);
    }
  }
  __syncthreads();

  // second GEMM: A = y1 (rows = neighbors, K = 64 hidden channels)
  v16bf a2[2][2];
#pragma unroll
  for (int mt = 0; mt < 2; ++mt)
#pragma unroll
    for (int ks = 0; ks < 2; ++ks)
      a2[mt][ks] = ld_frag_f32(&y1[wid][mt * 16 + l15][ks * 32 + 8 * half]);

#pragma unroll 1
  for (int ct = 0; ct < 4; ++ct) {
    const int col = ct * 16 + l15;
    v16bf bw[2];
#pragma unroll
    for (int ks = 0; ks < 2; ++ks)
      bw[ks] = ld_frag_bf(W2 + (size_t)col * 64 + ks * 32 + 8 * half);
    const float gg = g2[col], bb = b2[col];
    float pm = -INFINITY;
#pragma unroll
    for (int mt = 0; mt < 2; ++mt) {
      v8f acc = {};
#pragma unroll
      for (int ks = 0; ks < 2; ++ks)
        acc = __builtin_amdgcn_wmma_f32_16x16x32_bf16(false, a2[mt][ks], false, bw[ks],
                                                      (short)0, acc, false, false);
#pragma unroll
      for (int r = 0; r < 8; ++r) {
        // valid neighbor rows: mt0 -> 0..15; mt1 -> rows 16..19 (half==0, r<4)
        const bool rv = (mt == 0) || (half == 0 && r < 4);
        if (rv) pm = fmaxf(pm, leaky(acc[r] * gg + bb));
      }
    }
    pm = fmaxf(pm, __shfl_xor(pm, 16));     // combine row-halves (wave32)
    if (half == 0) out[(size_t)p * 64 + col] = (__bf16)pm;
  }
}

// ---------------------------------------------------------------------------
extern "C" void kernel_launch(void* const* d_in, const int* in_sizes, int n_in,
                              void* d_out, int out_size, void* d_ws, size_t ws_size,
                              hipStream_t stream) {
  const float* x    = (const float*)d_in[0];
  const float* w1   = (const float*)d_in[1];
  const float* g1   = (const float*)d_in[2];
  const float* b1   = (const float*)d_in[3];
  const float* w2   = (const float*)d_in[4];
  const float* g2   = (const float*)d_in[5];
  const float* b2   = (const float*)d_in[6];
  const float* wdg1 = (const float*)d_in[7];
  const float* gdg1 = (const float*)d_in[8];
  const float* bdg1 = (const float*)d_in[9];
  const float* wdg2 = (const float*)d_in[10];
  const float* gdg2 = (const float*)d_in[11];
  const float* bdg2 = (const float*)d_in[12];
  const float* wsn1 = (const float*)d_in[13];
  const float* gsn1 = (const float*)d_in[14];
  const float* bsn1 = (const float*)d_in[15];
  const float* wsn2 = (const float*)d_in[16];
  const float* gsn2 = (const float*)d_in[17];
  const float* bsn2 = (const float*)d_in[18];
  const float* w3   = (const float*)d_in[19];
  const float* g3   = (const float*)d_in[20];
  const float* b3   = (const float*)d_in[21];
  const float* w4   = (const float*)d_in[22];
  const float* g4   = (const float*)d_in[23];
  const float* b4   = (const float*)d_in[24];
  const float* w5   = (const float*)d_in[25];
  const float* g5   = (const float*)d_in[26];
  const float* b5   = (const float*)d_in[27];
  (void)ws_size; (void)in_sizes; (void)n_in; (void)out_size;

  const size_t P = (size_t)BATCH * NPTS;    // 32768 points

  char* wsp = (char*)d_ws;
  auto carve = [&](size_t bytes) -> void* {
    void* r = (void*)wsp;
    wsp += (bytes + 255) & ~(size_t)255;
    return r;
  };
  __bf16* h1b     = (__bf16*)carve(P * 64 * 2);
  __bf16* h2b     = (__bf16*)carve(P * 64 * 2);
  __bf16* hdgb    = (__bf16*)carve(P * 64 * 2);
  __bf16* hsnb    = (__bf16*)carve(P * 64 * 2);
  __bf16* h3b     = (__bf16*)carve(P * 64 * 2);
  __bf16* h4b     = (__bf16*)carve(P * 128 * 2);
  float*  xxv     = (float*)carve(P * 4);
  float*  basebuf = (float*)carve(P * 64 * 4);
  int*    idx1    = (int*)carve(P * KNN_K * 4);
  int*    idx2    = (int*)carve(P * KNN_K * 4);
  __bf16* w2bf    = (__bf16*)carve(64 * 64 * 2);
  __bf16* wdiffbf = (__bf16*)carve(64 * 64 * 2);
  __bf16* wbbf    = (__bf16*)carve(64 * 64 * 2);
  __bf16* wdg2bf  = (__bf16*)carve(64 * 64 * 2);
  __bf16* wsn1bf  = (__bf16*)carve(64 * 64 * 2);
  __bf16* wsn2bf  = (__bf16*)carve(64 * 64 * 2);
  __bf16* w3bf    = (__bf16*)carve(64 * 64 * 2);
  __bf16* w4bf    = (__bf16*)carve(128 * 64 * 2);
  __bf16* w5bf    = (__bf16*)carve(512 * 128 * 2);

  // weight preparation (tiny)
  k_cvt<<<16, 256, 0, stream>>>(w2, w2bf, 64 * 64);
  k_wprep<<<16, 256, 0, stream>>>(wdg1, wdiffbf, wbbf);
  k_cvt<<<16, 256, 0, stream>>>(wdg2, wdg2bf, 64 * 64);
  k_cvt<<<16, 256, 0, stream>>>(wsn1, wsn1bf, 64 * 64);
  k_cvt<<<16, 256, 0, stream>>>(wsn2, wsn2bf, 64 * 64);
  k_cvt<<<16, 256, 0, stream>>>(w3, w3bf, 64 * 64);
  k_cvt<<<32, 256, 0, stream>>>(w4, w4bf, 128 * 64);
  k_cvt<<<256, 256, 0, stream>>>(w5, w5bf, 512 * 128);

  // conv1 (3->64) + conv2 (64->64)
  k_conv1<<<(P * 64 + 255) / 256, 256, 0, stream>>>(x, w1, g1, b1, h1b);
  k_gemm<64, 64, true, false, false><<<P / 64, 128, 0, stream>>>(h1b, w2bf, g2, b2, h2b);

  // feature-space KNN (C=64) with fused Gram/top-k
  k_xx<<<(P + 255) / 256, 256, 0, stream>>>(h2b, xxv);
  k_knn<64><<<BATCH * (NPTS / 16), 128, 0, stream>>>(h2b, xxv, idx1);

  // dynamic-graph edge conv: base = h2 @ (Wa - Wb)^T, then gather-GEMM path
  k_gemm<64, 64, false, false, true><<<P / 64, 128, 0, stream>>>(h2b, wdiffbf,
                                                                 nullptr, nullptr, basebuf);
  k_edge<true><<<P / 4, 128, 0, stream>>>(h2b, idx1, basebuf, wbbf,
                                          gdg1, bdg1, wdg2bf, gdg2, bdg2, hdgb);

  // xyz-space KNN (C=3) + spatial-neighbor conv block
  k_knn<3><<<BATCH * (NPTS / 16), 128, 0, stream>>>(x, nullptr, idx2);
  k_edge<false><<<P / 4, 128, 0, stream>>>(hdgb, idx2, nullptr, wsn1bf,
                                           gsn1, bsn1, wsn2bf, gsn2, bsn2, hsnb);

  // final MLP 64 -> 64 -> 128 -> 512; conv5 writes (B,512,N,1) layout in f32
  k_gemm<64, 64, true, false, false><<<P / 64, 128, 0, stream>>>(hsnb, w3bf, g3, b3, h3b);
  k_gemm<64, 128, true, false, false><<<P / 64, 128, 0, stream>>>(h3b, w4bf, g4, b4, h4b);
  k_gemm<128, 512, true, true, true><<<P / 64, 128, 0, stream>>>(h4b, w5bf, g5, b5, d_out);
}
